// CNN_RNN_73701638799667
// MI455X (gfx1250) — matrix operand order repaired and checked by reading, not compiled, not run on hardware
//
#include <hip/hip_runtime.h>
#include <hip/hip_bf16.h>

typedef __attribute__((ext_vector_type(16))) __bf16 v16bf;
typedef __attribute__((ext_vector_type(8)))  float  v8f;

union Frag16 { v16bf v; unsigned u[8]; unsigned short h[16]; };

__device__ __forceinline__ unsigned short f2bf(float f) {
  union { float f; unsigned u; } c; c.f = f;
  unsigned u = c.u + 0x7fffu + ((c.u >> 16) & 1u);   // round-to-nearest-even
  return (unsigned short)(u >> 16);
}
__device__ __forceinline__ float bf2f(unsigned short h) {
  union { unsigned u; float f; } c; c.u = ((unsigned)h) << 16;
  return c.f;
}
__device__ __forceinline__ float sigf(float x) { return 1.0f / (1.0f + __expf(-x)); }

// ---- problem constants ----
constexpr int kFREQ = 258, kT = 2176, kNSH = 64;
constexpr int kH = 1960, kHP = 1984;          // GRU hidden, padded K
constexpr int kG3 = 5880, kG3P = 5888;        // 3*H, padded N
constexpr int kPOH1 = 84, kPOW1 = 41;         // conv1 pooled dims
constexpr int kPOH2 = 28, kPOW2 = 14;         // conv2 pooled dims

// ---------------------------------------------------------------------------
// init: zero feat (padded), zero hbf pad rows/cols, h = hx0, hbf = bf16(hx0)
// ---------------------------------------------------------------------------
__global__ void k_init(unsigned short* __restrict__ feat,
                       unsigned short* __restrict__ hbf,
                       float* __restrict__ h,
                       const float* __restrict__ hx0) {
  int idx = blockIdx.x * blockDim.x + threadIdx.x;
  const int FEATN = 128 * kHP;
  const int HBFN  = 16 * kHP;
  if (idx < FEATN) {
    feat[idx] = 0;
  } else if (idx < FEATN + HBFN) {
    int i = idx - FEATN;
    int row = i / kHP, col = i % kHP;
    float v = 0.f;
    if (row < 4 && col < kH) { v = hx0[row * kH + col]; h[row * kH + col] = v; }
    hbf[i] = f2bf(v);
  }
}

// ---------------------------------------------------------------------------
// convert GRU weights -> zero-padded bf16 [kG3P][kHP]
// ---------------------------------------------------------------------------
__global__ void k_convert(const float* __restrict__ wih, const float* __restrict__ whh,
                          unsigned short* __restrict__ wihb, unsigned short* __restrict__ whhb) {
  long long idx = (long long)blockIdx.x * blockDim.x + threadIdx.x;
  const long long NPER = (long long)kG3P * kHP;
  if (idx >= 2 * NPER) return;
  const float* src = (idx < NPER) ? wih : whh;
  unsigned short* dst = (idx < NPER) ? wihb : whhb;
  long long i = (idx < NPER) ? idx : idx - NPER;
  int row = (int)(i / kHP), col = (int)(i % kHP);
  float v = (row < kG3 && col < kH) ? src[(long long)row * kH + col] : 0.f;
  dst[i] = f2bf(v);
}

// ---------------------------------------------------------------------------
// conv1: implicit GEMM (M=144 pixels, N=64 ch, K=81->96) + bias + pool + sigmoid
// im2col staged once in LDS as bf16; A fragments = 2x ds_load_b128
// grid (3 pooled-col blocks, 84 pooled rows, 128 images), 128 threads
// ---------------------------------------------------------------------------
__global__ __launch_bounds__(128) void k_conv1(const float* __restrict__ x,
                                               const float* __restrict__ w1,
                                               const float* __restrict__ b1,
                                               unsigned short* __restrict__ pooled1) {
  __shared__ float patch[11][56];               //  2,464 B
  __shared__ unsigned short Bw[96][64];         // 12,288 B
  __shared__ unsigned short Acol[144 * 96];     // 27,648 B  (im2col, bf16)
  __shared__ unsigned short CpreH[144 * 64];    // 18,432 B  -> total 60,832 B

  const int bx  = blockIdx.x;
  const int poh = blockIdx.y;
  const int img = blockIdx.z;
  const int b = img >> 5, w = img & 31;
  const int oh0 = poh * 3;
  const int ow0 = bx * 48;
  const int tid = threadIdx.x;

  for (int i = tid; i < 11 * 56; i += 128) {
    int r = i / 56, c = i % 56;
    int ih = oh0 - 2 + r;          // valid 0..256 (row index into x[:,1:,:])
    int iw = ow0 - 2 + c;          // valid 0..127 (frame in window)
    float v = 0.f;
    if (ih >= 0 && ih < 257 && iw >= 0 && iw < 128)
      v = x[(size_t)b * kFREQ * kT + (size_t)(ih + 1) * kT + (w * kNSH + iw)];
    patch[r][c] = v;
  }
  for (int i = tid; i < 96 * 64; i += 128) {
    int k = i >> 6, oc = i & 63;
    Bw[k][oc] = (k < 81) ? f2bf(w1[oc * 81 + k]) : (unsigned short)0;
  }
  __syncthreads();

  // build bf16 im2col: Acol[pix][k], k padded 81->96
  for (int i = tid; i < 144 * 96; i += 128) {
    int pix = i / 96, k = i % 96;
    int ohl = pix / 48, owl = pix % 48;
    unsigned short v = 0;
    if (k < 81) {
      int ky = k / 9, kx = k - 9 * ky;
      v = f2bf(patch[ohl + ky][owl + kx]);
    }
    Acol[i] = v;
  }
  __syncthreads();

  const int wave = tid >> 5, lane = tid & 31;
  const int ocB = wave * 16;
  const int lm = lane & 15, hiL = lane >> 4;

  Frag16 bfr[3];                                  // hoisted B fragments
  for (int kc = 0; kc < 3; ++kc) {
    int kb = kc * 32 + hiL * 16;
    for (int v = 0; v < 8; ++v) {
      unsigned short h0 = Bw[kb + 2 * v + 0][ocB + lm];
      unsigned short h1 = Bw[kb + 2 * v + 1][ocB + lm];
      bfr[kc].u[v] = (unsigned)h0 | ((unsigned)h1 << 16);
    }
  }

  for (int mt = 0; mt < 9; ++mt) {
    v8f acc = {};
    const int pix = mt * 16 + lm;
    const unsigned short* arow = &Acol[pix * 96];
    for (int kc = 0; kc < 3; ++kc) {
      Frag16 af;
      const int abase = kc * 32 + hiL * 16;
      uint4 alo = *(const uint4*)(arow + abase);       // ds_load_b128
      uint4 ahi = *(const uint4*)(arow + abase + 8);   // ds_load_b128
      af.u[0] = alo.x; af.u[1] = alo.y; af.u[2] = alo.z; af.u[3] = alo.w;
      af.u[4] = ahi.x; af.u[5] = ahi.y; af.u[6] = ahi.z; af.u[7] = ahi.w;
      acc = __builtin_amdgcn_wmma_f32_16x16x32_bf16(false, af.v, false, bfr[kc].v,
                                                    (short)0, acc, false, false);
    }
    for (int r = 0; r < 8; ++r)
      CpreH[(mt * 16 + r + hiL * 8) * 64 + (ocB + lm)] = f2bf(acc[r]);
  }
  __syncthreads();

  for (int i = tid; i < 64 * 16; i += 128) {      // fused bias + 3x3 pool + sigmoid
    int ch = i >> 4, pc = i & 15;
    int pw = bx * 16 + pc;
    if (pw >= kPOW1) continue;
    float m = -3.0e38f;
    for (int dr = 0; dr < 3; ++dr)
      for (int dc = 0; dc < 3; ++dc)
        m = fmaxf(m, bf2f(CpreH[(dr * 48 + pc * 3 + dc) * 64 + ch]));
    float val = sigf(m + b1[ch]);
    pooled1[(((size_t)img * 64 + ch) * kPOH1 + poh) * kPOW1 + pw] = f2bf(val);
  }
}

// ---------------------------------------------------------------------------
// conv2: implicit GEMM (M=126 pixels, N=5->16, K=1024) + bias + pool + sigmoid
// per-chunk im2col slice in LDS; A and B fragments = 2x ds_load_b128 each
// grid (28 pooled rows, 128 images), 128 threads
// ---------------------------------------------------------------------------
__global__ __launch_bounds__(128) void k_conv2(const unsigned short* __restrict__ pooled1,
                                               const float* __restrict__ w2,
                                               const float* __restrict__ b2,
                                               unsigned short* __restrict__ feat) {
  __shared__ unsigned short patch[64][6][45];   // 34,560 B (bf16)
  __shared__ unsigned short Bwt[5][1024];       // 10,240 B (n-major weights)
  __shared__ unsigned short chunkA[128 * 32];   //  8,192 B (per-chunk im2col)
  __shared__ float Cpre[128][16];               //  8,192 B -> total 61,184 B

  const int poh = blockIdx.x;
  const int img = blockIdx.y;
  const int b = img >> 5, w = img & 31;
  const int oh0 = poh * 3;
  const int tid = threadIdx.x;

  for (int i = tid; i < 64 * 6 * 45; i += 128) {
    int ic = i / 270;
    int rem = i - ic * 270;
    int r = rem / 45, c = rem % 45;
    int ih = oh0 - 2 + r;          // valid 0..83
    int iw = c - 2;                // valid 0..40
    unsigned short v = 0;
    if (ih >= 0 && ih < kPOH1 && iw >= 0 && iw < kPOW1)
      v = pooled1[(((size_t)img * 64 + ic) * kPOH1 + ih) * kPOW1 + iw];
    patch[ic][r][c] = v;
  }
  for (int i = tid; i < 5 * 1024; i += 128) {
    int oc = i >> 10, k = i & 1023;
    int ic = k >> 4, kk = k & 15, ky = kk >> 2, kx = kk & 3;
    Bwt[oc][k] = f2bf(w2[(((oc * 64) + ic) * 4 + ky) * 4 + kx]);
  }

  const int wave = tid >> 5, lane = tid & 31;
  const int lm = lane & 15, hiL = lane >> 4;
  const int mt0 = wave * 2;

  // per-thread pixel decomposition for the chunk-A build (pix == tid)
  const int myohl = tid / 42, myowl = tid % 42;
  const bool myvalid = (tid < 126);

  v8f acc0 = {}, acc1 = {};
  for (int kc = 0; kc < 32; ++kc) {
    __syncthreads();   // patch/Bwt ready (first iter) / frags consumed (later)
    // each thread builds the 32-wide K-slice of its own pixel row
    unsigned short* arow = &chunkA[tid * 32];
    if (myvalid) {
      for (int kk = 0; kk < 32; ++kk) {
        int k = kc * 32 + kk;
        int ic = k >> 4, kr = k & 15, ky = kr >> 2, kx = kr & 3;
        arow[kk] = patch[ic][myohl + ky][myowl + kx];
      }
    } else {
      for (int kk = 0; kk < 32; ++kk) arow[kk] = 0;
    }
    __syncthreads();

    Frag16 bf;
    if (lm < 5) {
      const unsigned short* brow = &Bwt[lm][kc * 32 + hiL * 16];
      uint4 b0 = *(const uint4*)(brow);        // ds_load_b128
      uint4 b1 = *(const uint4*)(brow + 8);    // ds_load_b128
      bf.u[0] = b0.x; bf.u[1] = b0.y; bf.u[2] = b0.z; bf.u[3] = b0.w;
      bf.u[4] = b1.x; bf.u[5] = b1.y; bf.u[6] = b1.z; bf.u[7] = b1.w;
    } else {
      for (int v = 0; v < 8; ++v) bf.u[v] = 0;
    }

    for (int j = 0; j < 2; ++j) {
      const int pix = (mt0 + j) * 16 + lm;
      const unsigned short* arow2 = &chunkA[pix * 32 + hiL * 16];
      Frag16 af;
      uint4 alo = *(const uint4*)(arow2);       // ds_load_b128
      uint4 ahi = *(const uint4*)(arow2 + 8);   // ds_load_b128
      af.u[0] = alo.x; af.u[1] = alo.y; af.u[2] = alo.z; af.u[3] = alo.w;
      af.u[4] = ahi.x; af.u[5] = ahi.y; af.u[6] = ahi.z; af.u[7] = ahi.w;
      if (j == 0)
        acc0 = __builtin_amdgcn_wmma_f32_16x16x32_bf16(false, af.v, false, bf.v,
                                                       (short)0, acc0, false, false);
      else
        acc1 = __builtin_amdgcn_wmma_f32_16x16x32_bf16(false, af.v, false, bf.v,
                                                       (short)0, acc1, false, false);
    }
  }
  __syncthreads();
  for (int r = 0; r < 8; ++r) {
    Cpre[mt0 * 16 + r + hiL * 8][lm]       = acc0[r];
    Cpre[(mt0 + 1) * 16 + r + hiL * 8][lm] = acc1[r];
  }
  __syncthreads();

  for (int i = tid; i < 5 * kPOW2; i += 128) {
    int oc = i / kPOW2, pw = i % kPOW2;
    float m = -3.0e38f;
    for (int dr = 0; dr < 3; ++dr)
      for (int dc = 0; dc < 3; ++dc)
        m = fmaxf(m, Cpre[dr * 42 + pw * 3 + dc][oc]);
    float val = sigf(m + b2[oc]);
    int hidx = oc * (kPOH2 * kPOW2) + poh * kPOW2 + pw;
    feat[(size_t)(w * 4 + b) * kHP + hidx] = f2bf(val);
  }
}

// ---------------------------------------------------------------------------
// gi = feat(128 x 1984) @ w_ih^T  -> (128 x 5888), all timesteps batched
// grid (368 N-tiles, 2), 128 threads; 1 WMMA tile per wave, K loop of 62
// ---------------------------------------------------------------------------
__global__ __launch_bounds__(128) void k_gemm_gi(const unsigned short* __restrict__ feat,
                                                 const unsigned short* __restrict__ wb,
                                                 float* __restrict__ gi) {
  const int tid = threadIdx.x;
  const int wave = tid >> 5, lane = tid & 31;
  const int lm = lane & 15, hiL = lane >> 4;
  const int nt = blockIdx.x;
  const int mt = blockIdx.y * 4 + wave;
  const int n = nt * 16 + lm;
  const unsigned short* ap = feat + (size_t)(mt * 16 + lm) * kHP;
  const unsigned short* bp = wb + (size_t)n * kHP;
  v8f acc = {};
  for (int kc = 0; kc < kHP / 32; ++kc) {
    Frag16 af, bf;
    int abase = kc * 32 + hiL * 16;
    uint4 alo = *(const uint4*)(ap + abase);
    uint4 ahi = *(const uint4*)(ap + abase + 8);
    af.u[0] = alo.x; af.u[1] = alo.y; af.u[2] = alo.z; af.u[3] = alo.w;
    af.u[4] = ahi.x; af.u[5] = ahi.y; af.u[6] = ahi.z; af.u[7] = ahi.w;
    int bbase = kc * 32 + hiL * 16;
    uint4 b0 = *(const uint4*)(bp + bbase);
    uint4 b1 = *(const uint4*)(bp + bbase + 8);
    bf.u[0] = b0.x; bf.u[1] = b0.y; bf.u[2] = b0.z; bf.u[3] = b0.w;
    bf.u[4] = b1.x; bf.u[5] = b1.y; bf.u[6] = b1.z; bf.u[7] = b1.w;
    __builtin_prefetch((const void*)(bp + bbase + 128), 0, 1);
    acc = __builtin_amdgcn_wmma_f32_16x16x32_bf16(false, af.v, false, bf.v,
                                                  (short)0, acc, false, false);
  }
  for (int r = 0; r < 8; ++r)
    gi[(size_t)(mt * 16 + r + hiL * 8) * kG3P + n] = acc[r];
}

// ---------------------------------------------------------------------------
// gh = h(4 x 1960, padded to 16 rows) @ w_hh^T -> (16 x 5888)
// grid 92 blocks x 4 waves = 368 N-tiles
// ---------------------------------------------------------------------------
__global__ __launch_bounds__(128) void k_gemm_gh(const unsigned short* __restrict__ hbf,
                                                 const unsigned short* __restrict__ wb,
                                                 float* __restrict__ gh) {
  const int tid = threadIdx.x;
  const int wave = tid >> 5, lane = tid & 31;
  const int lm = lane & 15, hiL = lane >> 4;
  const int nt = blockIdx.x * 4 + wave;
  const int n = nt * 16 + lm;
  const unsigned short* ap = hbf + (size_t)lm * kHP;
  const unsigned short* bp = wb + (size_t)n * kHP;
  v8f acc = {};
  for (int kc = 0; kc < kHP / 32; ++kc) {
    Frag16 af, bf;
    int abase = kc * 32 + hiL * 16;
    uint4 alo = *(const uint4*)(ap + abase);
    uint4 ahi = *(const uint4*)(ap + abase + 8);
    af.u[0] = alo.x; af.u[1] = alo.y; af.u[2] = alo.z; af.u[3] = alo.w;
    af.u[4] = ahi.x; af.u[5] = ahi.y; af.u[6] = ahi.z; af.u[7] = ahi.w;
    int bbase = kc * 32 + hiL * 16;
    uint4 b0 = *(const uint4*)(bp + bbase);
    uint4 b1 = *(const uint4*)(bp + bbase + 8);
    bf.u[0] = b0.x; bf.u[1] = b0.y; bf.u[2] = b0.z; bf.u[3] = b0.w;
    bf.u[4] = b1.x; bf.u[5] = b1.y; bf.u[6] = b1.z; bf.u[7] = b1.w;
    __builtin_prefetch((const void*)(bp + bbase + 128), 0, 1);
    acc = __builtin_amdgcn_wmma_f32_16x16x32_bf16(false, af.v, false, bf.v,
                                                  (short)0, acc, false, false);
  }
  for (int r = 0; r < 8; ++r)
    gh[(size_t)(r + hiL * 8) * kG3P + n] = acc[r];
}

// ---------------------------------------------------------------------------
// GRU elementwise update for step t
// ---------------------------------------------------------------------------
__global__ void k_gru_update(const float* __restrict__ gi, const float* __restrict__ gh,
                             const float* __restrict__ bih, const float* __restrict__ bhh,
                             float* __restrict__ h, unsigned short* __restrict__ hbf, int t) {
  int idx = blockIdx.x * blockDim.x + threadIdx.x;
  if (idx >= 4 * kH) return;
  int b = idx / kH, j = idx % kH;
  const float* gir = gi + (size_t)(t * 4 + b) * kG3P;
  const float* ghr = gh + (size_t)b * kG3P;
  float ir = gir[j]          + bih[j];
  float iz = gir[j + kH]     + bih[j + kH];
  float in = gir[j + 2 * kH] + bih[j + 2 * kH];
  float hr = ghr[j]          + bhh[j];
  float hz = ghr[j + kH]     + bhh[j + kH];
  float hn = ghr[j + 2 * kH] + bhh[j + 2 * kH];
  float r = sigf(ir + hr);
  float z = sigf(iz + hz);
  float nn = tanhf(in + r * hn);
  float hnew = (1.f - z) * nn + z * h[idx];
  h[idx] = hnew;
  hbf[(size_t)b * kHP + j] = f2bf(hnew);
}

// ---------------------------------------------------------------------------
// head: logits = h @ fc_w^T + fc_b, softmax; write samples then labels
// ---------------------------------------------------------------------------
__global__ void k_head(const float* __restrict__ h, const float* __restrict__ fcw,
                       const float* __restrict__ fcb, const int* __restrict__ labels,
                       float* __restrict__ out, int out_size) {
  __shared__ float logits[8];
  int tid = threadIdx.x;
  int pair = tid >> 5, lane = tid & 31;
  int b = pair >> 1, c = pair & 1;
  float s = 0.f;
  for (int j = lane; j < kH; j += 32)
    s += h[b * kH + j] * fcw[c * kH + j];
  for (int off = 16; off; off >>= 1)
    s += __shfl_down(s, off, 32);
  if (lane == 0) logits[pair] = s + fcb[c];
  __syncthreads();
  if (tid < 4) {
    float l0 = logits[tid * 2], l1 = logits[tid * 2 + 1];
    float m = fmaxf(l0, l1);
    float e0 = __expf(l0 - m), e1 = __expf(l1 - m);
    float inv = 1.f / (e0 + e1);
    if (out_size > tid * 2 + 1) { out[tid * 2] = e0 * inv; out[tid * 2 + 1] = e1 * inv; }
    if (out_size > 8 + tid) out[8 + tid] = (float)labels[tid];
  }
}

extern "C" void kernel_launch(void* const* d_in, const int* in_sizes, int n_in,
                              void* d_out, int out_size, void* d_ws, size_t ws_size,
                              hipStream_t stream) {
  const float* x     = (const float*)d_in[0];
  const float* hx0   = (const float*)d_in[1];
  const int*   labels = (const int*)d_in[2];
  const float* c1w   = (const float*)d_in[3];
  const float* c1b   = (const float*)d_in[4];
  const float* c2w   = (const float*)d_in[5];
  const float* c2b   = (const float*)d_in[6];
  const float* wih   = (const float*)d_in[7];
  const float* whh   = (const float*)d_in[8];
  const float* bih   = (const float*)d_in[9];
  const float* bhh   = (const float*)d_in[10];
  const float* fcw   = (const float*)d_in[11];
  const float* fcb   = (const float*)d_in[12];
  (void)in_sizes; (void)n_in; (void)ws_size;

  char* base = (char*)d_ws;
  size_t off = 0;
  auto carve = [&](size_t bytes) {
    char* p = base + off;
    off = (off + bytes + 255) & ~(size_t)255;
    return p;
  };
  unsigned short* pooled1 = (unsigned short*)carve((size_t)128 * 64 * kPOH1 * kPOW1 * 2);
  unsigned short* wihb    = (unsigned short*)carve((size_t)kG3P * kHP * 2);
  unsigned short* whhb    = (unsigned short*)carve((size_t)kG3P * kHP * 2);
  unsigned short* feat    = (unsigned short*)carve((size_t)128 * kHP * 2);
  float*          gi      = (float*)carve((size_t)128 * kG3P * 4);
  float*          gh      = (float*)carve((size_t)16 * kG3P * 4);
  float*          h       = (float*)carve((size_t)4 * kH * 4);
  unsigned short* hbf     = (unsigned short*)carve((size_t)16 * kHP * 2);

  k_init<<<(128 * kHP + 16 * kHP + 255) / 256, 256, 0, stream>>>(feat, hbf, h, hx0);
  {
    long long n = 2LL * kG3P * kHP;
    k_convert<<<(unsigned)((n + 255) / 256), 256, 0, stream>>>(wih, whh, wihb, whhb);
  }
  k_conv1<<<dim3(3, 84, 128), 128, 0, stream>>>(x, c1w, c1b, pooled1);
  k_conv2<<<dim3(28, 128), 128, 0, stream>>>(pooled1, c2w, c2b, feat);
  k_gemm_gi<<<dim3(368, 2), 128, 0, stream>>>(feat, wihb, gi);
  for (int t = 0; t < 32; ++t) {
    k_gemm_gh<<<92, 128, 0, stream>>>(hbf, whhb, gh);
    k_gru_update<<<(4 * kH + 255) / 256, 256, 0, stream>>>(gi, gh, bih, bhh, h, hbf, t);
  }
  k_head<<<1, 256, 0, stream>>>(h, fcw, fcb, labels, (float*)d_out, out_size);
}
